// CausalMultiHeadGroupAttention_89309549953268
// MI455X (gfx1250) — compile-verified, matrix operand order checked by reading, not run on hardware
//
#include <hip/hip_runtime.h>

// ---------------------------------------------------------------------------
// CDNA5 (gfx1250) causal GQA attention block.
// bf16 WMMA + f32 accumulate; bf16 intermediates; b128 LDS fragment loads;
// async global->LDS staging (GLOBAL_LOAD_ASYNC_TO_LDS_B128);
// v_perm-packed b32 transposed staging for the K-contiguous LDS layouts.
// ---------------------------------------------------------------------------

typedef __attribute__((ext_vector_type(16))) __bf16 v16bf;
typedef __attribute__((ext_vector_type(8)))  __bf16 v8bf;
typedef __attribute__((ext_vector_type(8)))  float  v8f;
typedef __attribute__((ext_vector_type(4)))  int    v4i;
typedef __attribute__((ext_vector_type(4)))  unsigned u4;

#define BATCH 2
#define SEQ   2048
#define EMB   2048
#define NH    16
#define NKV   4
#define HD    128
#define GRP   (NH / NKV)

#if defined(__has_builtin)
#if __has_builtin(__builtin_amdgcn_global_load_async_to_lds_b128)
#define HAS_ASYNC_B128 1
#endif
#if __has_builtin(__builtin_amdgcn_s_wait_asynccnt)
#define HAS_WAIT_ASYNC 1
#endif
#endif

// float -> bf16 bits, round-to-nearest-even
static __device__ __forceinline__ unsigned short f2bf_u16(float f) {
  union { float f; unsigned u; } x; x.f = f;
  unsigned r = x.u + 0x7FFFu + ((x.u >> 16) & 1u);
  return (unsigned short)(r >> 16);
}
static __device__ __forceinline__ float bf2f(unsigned short h) {
  union { unsigned u; float f; } x; x.u = ((unsigned)h) << 16;
  return x.f;
}

static __device__ __forceinline__ v8f wmma_bf16(v16bf a, v16bf b, v8f c) {
  return __builtin_amdgcn_wmma_f32_16x16x32_bf16(false, a, false, b, (short)0, c,
                                                 false, false);
}

// Build a 16x32 A/B fragment whose K-dim is contiguous at p:
// i<8 -> k = kb+i (shorts p[0..7]); i>=8 -> k = kb+16+(i-8) (shorts p[16..23]).
// Two aligned 16B loads + concat. Caller passes p = &buf[... + kb].
static __device__ __forceinline__ v16bf ldfrag(const unsigned short* p) {
  v8bf lo = *(const v8bf*)p;
  v8bf hi = *(const v8bf*)(p + 16);
  return __builtin_shufflevector(lo, hi, 0, 1, 2, 3, 4, 5, 6, 7,
                                 8, 9, 10, 11, 12, 13, 14, 15);
}

// 16-byte global -> LDS copy; async on gfx1250.
static __device__ __forceinline__ void cp16(void* lds, const void* g) {
#ifdef HAS_ASYNC_B128
  __builtin_amdgcn_global_load_async_to_lds_b128(
      (__attribute__((address_space(1))) v4i*)(unsigned long long)g,
      (__attribute__((address_space(3))) v4i*)(unsigned)(unsigned long long)lds,
      0, 0);
#else
  *(uint4*)lds = *(const uint4*)g;
#endif
}
static __device__ __forceinline__ void cp16_fence() {
#ifdef HAS_WAIT_ASYNC
  __builtin_amdgcn_s_wait_asynccnt(0);
#elif defined(HAS_ASYNC_B128)
  asm volatile("s_wait_asynccnt 0" ::: "memory");
#endif
}

// Transpose-stage two k-rows x 8 cols: pack (rowK[c], rowK1[c]) into one u32
// with v_perm_b32 and store b32 at s[(c8+j)*stride + 2*r2] (4B aligned).
static __device__ __forceinline__ void stage_t2(unsigned short* sbase,
                                                int stride_sh,
                                                const unsigned short* g0,
                                                const unsigned short* g1,
                                                int c8, int r2) {
  u4 a0 = *(const u4*)g0;
  u4 a1 = *(const u4*)g1;
#pragma unroll
  for (int j = 0; j < 8; ++j) {
    unsigned lo = a0[j >> 1], hi = a1[j >> 1];
    unsigned sel = (j & 1) ? 0x07060302u : 0x05040100u;
    unsigned pk = __builtin_amdgcn_perm(hi, lo, sel);
    *(unsigned*)&sbase[(c8 + j) * stride_sh + 2 * r2] = pk;
  }
}

// ---------------------------------------------------------------------------
// f32 -> bf16 buffer conversion (done once per input buffer)
// ---------------------------------------------------------------------------
__global__ void f32_to_bf16_kernel(const float* __restrict__ in,
                                   unsigned short* __restrict__ out, int n) {
  int i = blockIdx.x * blockDim.x + threadIdx.x;
  if (i < n) out[i] = f2bf_u16(in[i]);
}

// ---------------------------------------------------------------------------
// GEMM: C[M,N] = A[M,K]*W[K,N], A/W bf16 row-major. Block tile 128x128, BK=64,
// 256 threads = 8 waves (4 along M x 2 along N), each wave 32x64 via 2x4 WMMA.
// A tile async-staged; W tile staged transposed (packed b32 stores) so both
// operands' K-dim is contiguous in LDS (pure b128 fragment loads).
// ---------------------------------------------------------------------------
template <bool OUT_BF16>
__global__ __launch_bounds__(256)
void gemm_bf16_wmma(const unsigned short* __restrict__ A,
                    const unsigned short* __restrict__ W,
                    const float* __restrict__ bias, void* __restrict__ Cv,
                    int M, int N, int K) {
  __shared__ unsigned short sA[128 * 64];    // [row][k], k contiguous, 16B rows
  __shared__ unsigned short sBt[128 * 72];   // [col][k], stride 72 (16B mult)

  const int tid  = threadIdx.x;
  const int lane = tid & 31;
  const int wid  = tid >> 5;
  const int wm   = wid & 3;
  const int wn   = wid >> 2;
  const int row0 = blockIdx.y * 128;
  const int col0 = blockIdx.x * 128;
  const int l15  = lane & 15;
  const int kb   = (lane & 16) ? 8 : 0;
  const int hi   = (lane & 16) ? 8 : 0;

  v8f acc[2][4];
#pragma unroll
  for (int mt = 0; mt < 2; ++mt)
#pragma unroll
    for (int nt = 0; nt < 4; ++nt) acc[mt][nt] = (v8f){};

  for (int k0 = 0; k0 < K; k0 += 64) {
    // A tile: 128x64 bf16 = 1024 x 16B chunks, async copy
#pragma unroll
    for (int t = 0; t < 4; ++t) {
      int cid = tid + t * 256;
      int r = cid >> 3, c8 = (cid & 7) * 8;
      cp16(&sA[r * 64 + c8], A + (size_t)(row0 + r) * K + k0 + c8);
    }
    // W tile: 64x128 bf16, transposed into sBt[col][k], packed pair stores
#pragma unroll
    for (int t = 0; t < 2; ++t) {
      int cid = tid + t * 256;              // 512 pair-chunks
      int r2 = cid >> 4, c8 = (cid & 15) * 8;
      const unsigned short* g0 = W + (size_t)(k0 + 2 * r2) * N + col0 + c8;
      stage_t2(sBt, 72, g0, g0 + N, c8, r2);
    }
    cp16_fence();
    __syncthreads();

#pragma unroll
    for (int kk = 0; kk < 64; kk += 32) {
      v16bf afr[2], bfr[4];
#pragma unroll
      for (int mt = 0; mt < 2; ++mt)
        afr[mt] = ldfrag(&sA[(wm * 32 + mt * 16 + l15) * 64 + kk + kb]);
#pragma unroll
      for (int nt = 0; nt < 4; ++nt)
        bfr[nt] = ldfrag(&sBt[(wn * 64 + nt * 16 + l15) * 72 + kk + kb]);
#pragma unroll
      for (int mt = 0; mt < 2; ++mt)
#pragma unroll
        for (int nt = 0; nt < 4; ++nt)
          acc[mt][nt] = wmma_bf16(afr[mt], bfr[nt], acc[mt][nt]);
    }
    __syncthreads();
  }

#pragma unroll
  for (int mt = 0; mt < 2; ++mt)
#pragma unroll
    for (int nt = 0; nt < 4; ++nt)
#pragma unroll
      for (int r = 0; r < 8; ++r) {
        int row = row0 + wm * 32 + mt * 16 + r + hi;
        int col = col0 + wn * 64 + nt * 16 + l15;
        float vo = acc[mt][nt][r];
        if (OUT_BF16) {
          ((unsigned short*)Cv)[(size_t)row * N + col] = f2bf_u16(vo);
        } else {
          ((float*)Cv)[(size_t)row * N + col] = vo + bias[col];
        }
      }
}

// ---------------------------------------------------------------------------
// RoPE (reference's sin-only variant), in place over bf16 [B,S,nh,HD].
// ---------------------------------------------------------------------------
__global__ void rope_kernel(unsigned short* __restrict__ t, int nh, int total) {
  int idx = blockIdx.x * blockDim.x + threadIdx.x;
  if (idx >= total) return;
  int j = idx & 63;
  int rest = idx >> 6;
  int h = rest % nh; rest /= nh;
  int s = rest % SEQ;
  int b = rest / SEQ;
  float theta = __expf(-(float)j * 0.28782313662425575f);  // ln(1e4)/32
  float sv = __sinf((float)s * theta);
  unsigned short* p = t + (((size_t)(b * SEQ + s) * nh + h) * HD);
  float xr = bf2f(p[j]), xi = bf2f(p[j + 64]);
  p[j]      = f2bf_u16(xr * sv - xi * sv);
  p[j + 64] = f2bf_u16(xi * sv + xr * sv);
}

// ---------------------------------------------------------------------------
// Flash attention, causal, GQA, all-bf16 streams. Grid (S/128, NH, B).
// 8 waves x 16 q-rows. K chunk (64x128) async-staged [kv][d]; V staged
// transposed [d][kv] (packed b32 stores); P restaged through the dead K
// region for the PV A-fragments.
// ---------------------------------------------------------------------------
__global__ __launch_bounds__(256)
void attn_kernel(const unsigned short* __restrict__ qg,
                 const unsigned short* __restrict__ kgp,
                 const unsigned short* __restrict__ vgp,
                 unsigned short* __restrict__ yg) {
  // union: K tile [64][128] (8192 u16)  OR  P [8 waves][16][72] (9216 u16)
  __shared__ unsigned short sKP[9216];
  __shared__ unsigned short sVt[128 * 72];  // [d][kv], stride 72

  const int tid  = threadIdx.x;
  const int lane = tid & 31;
  const int wid  = tid >> 5;
  const int l15  = lane & 15;
  const int kb   = (lane & 16) ? 8 : 0;
  const int hi   = (lane & 16) ? 8 : 0;
  const int qt   = blockIdx.x;
  const int h    = blockIdx.y;
  const int b    = blockIdx.z;
  const int g    = h / GRP;
  const int qrow0 = qt * 128 + wid * 16;
  const float scale = 0.08838834764831845f;  // 1/sqrt(128)

  // Q fragments: 4 chunks of 32 along head_dim, pure b128 global loads
  v16bf Qf[4];
  const size_t qoff = (size_t)(b * SEQ + qrow0 + l15) * (NH * HD) + h * HD;
#pragma unroll
  for (int kc = 0; kc < 4; ++kc) Qf[kc] = ldfrag(qg + qoff + kc * 32 + kb);

  v8f O[8];
#pragma unroll
  for (int nt = 0; nt < 8; ++nt) O[nt] = (v8f){};
  float mrow[8], lrow[8];
#pragma unroll
  for (int r = 0; r < 8; ++r) { mrow[r] = -3.0e38f; lrow[r] = 0.0f; }

  const int nchunks = qt * 2 + 2;
  for (int c = 0; c < nchunks; ++c) {
    const int kv0 = c * 64;
    // K tile 64x128: async b128 copies into [kv][d]
#pragma unroll
    for (int t = 0; t < 4; ++t) {
      int cid = tid + t * 256;
      int r = cid >> 4, c8 = (cid & 15) * 8;
      cp16(&sKP[r * 128 + c8],
           kgp + ((size_t)(b * SEQ + kv0 + r) * NKV + g) * HD + c8);
    }
    // V tile 64x128 -> transposed [d][kv], packed pair stores
#pragma unroll
    for (int t = 0; t < 2; ++t) {
      int cid = tid + t * 256;              // 512 pair-chunks
      int r2 = cid >> 4, c8 = (cid & 15) * 8;
      const unsigned short* g0 =
          vgp + ((size_t)(b * SEQ + kv0 + 2 * r2) * NKV + g) * HD + c8;
      stage_t2(sVt, 72, g0, g0 + (size_t)NKV * HD, c8, r2);
    }
    cp16_fence();
    __syncthreads();

    // scores S(16x64) = Q(16x128) * K^T
    v8f Sa[4];
#pragma unroll
    for (int nt = 0; nt < 4; ++nt) Sa[nt] = (v8f){};
#pragma unroll
    for (int kc = 0; kc < 4; ++kc)
#pragma unroll
      for (int nt = 0; nt < 4; ++nt) {
        v16bf bf = ldfrag(&sKP[(nt * 16 + l15) * 128 + kc * 32 + kb]);
        Sa[nt] = wmma_bf16(Qf[kc], bf, Sa[nt]);
      }
    __syncthreads();  // all waves done with K tile; safe to alias with P

    // online softmax; write P into this wave's private region
#pragma unroll
    for (int r = 0; r < 8; ++r) {
      int qrow = qrow0 + r + hi;
      float sv4[4];
      float smax = -3.0e38f;
#pragma unroll
      for (int nt = 0; nt < 4; ++nt) {
        int col = kv0 + nt * 16 + l15;
        float s = Sa[nt][r] * scale;
        if (col > qrow) s = -3.0e38f;
        sv4[nt] = s;
        smax = fmaxf(smax, s);
      }
#pragma unroll
      for (int m = 8; m >= 1; m >>= 1)
        smax = fmaxf(smax, __shfl_xor(smax, m, 32));
      float mnew  = fmaxf(mrow[r], smax);
      float alpha = __expf(mrow[r] - mnew);
      float psum = 0.0f;
#pragma unroll
      for (int nt = 0; nt < 4; ++nt) {
        float p = __expf(sv4[nt] - mnew);
        psum += p;
        sKP[wid * 1152 + (r + hi) * 72 + nt * 16 + l15] = f2bf_u16(p);
      }
#pragma unroll
      for (int m = 8; m >= 1; m >>= 1) psum += __shfl_xor(psum, m, 32);
      lrow[r] = lrow[r] * alpha + psum;
      mrow[r] = mnew;
#pragma unroll
      for (int nt = 0; nt < 8; ++nt) O[nt][r] *= alpha;
    }
    // own-region P readback: same-wave DS ops are in order, no barrier needed

    // O(16x128) += P(16x64) * V(64x128)
#pragma unroll
    for (int pk = 0; pk < 2; ++pk) {
      v16bf af = ldfrag(&sKP[wid * 1152 + l15 * 72 + pk * 32 + kb]);
#pragma unroll
      for (int nt = 0; nt < 8; ++nt) {
        v16bf bf = ldfrag(&sVt[(nt * 16 + l15) * 72 + pk * 32 + kb]);
        O[nt] = wmma_bf16(af, bf, O[nt]);
      }
    }
    __syncthreads();  // done with sVt / sKP before next chunk's staging
  }

  // normalize + store bf16: y[b, qrow, h, :]
#pragma unroll
  for (int nt = 0; nt < 8; ++nt)
#pragma unroll
    for (int r = 0; r < 8; ++r) {
      int row = qrow0 + r + hi;
      int d = nt * 16 + l15;
      yg[(size_t)(b * SEQ + row) * (NH * HD) + h * HD + d] =
          f2bf_u16(O[nt][r] / lrow[r]);
    }
}

// ---------------------------------------------------------------------------
// Launch. Workspace (bf16 u16 elements):
//   xb 8.39M | Wqb 4.19M | Wkb 1.05M | Wvb 1.05M | Wob 4.19M |
//   qb 8.39M | kb 2.10M | vb 2.10M | yb 8.39M   (~80 MB total)
// ---------------------------------------------------------------------------
extern "C" void kernel_launch(void* const* d_in, const int* in_sizes, int n_in,
                              void* d_out, int out_size, void* d_ws, size_t ws_size,
                              hipStream_t stream) {
  const float* x  = (const float*)d_in[0];
  const float* Wq = (const float*)d_in[1];
  const float* Wk = (const float*)d_in[2];
  const float* Wv = (const float*)d_in[3];
  const float* Wo = (const float*)d_in[4];
  const float* bo = (const float*)d_in[5];
  float* out = (float*)d_out;

  unsigned short* ws = (unsigned short*)d_ws;
  unsigned short* xb  = ws;
  unsigned short* wqb = xb  + (size_t)BATCH * SEQ * EMB;
  unsigned short* wkb = wqb + (size_t)EMB * (NH * HD);
  unsigned short* wvb = wkb + (size_t)EMB * (NKV * HD);
  unsigned short* wob = wvb + (size_t)EMB * (NKV * HD);
  unsigned short* qb  = wob + (size_t)(NH * HD) * (NH * HD);
  unsigned short* kb  = qb  + (size_t)BATCH * SEQ * NH * HD;
  unsigned short* vb  = kb  + (size_t)BATCH * SEQ * NKV * HD;
  unsigned short* yb  = vb  + (size_t)BATCH * SEQ * NKV * HD;

  const int M = BATCH * SEQ;  // 4096
  dim3 blk(256);

  // one-time bf16 conversions
  int nx = BATCH * SEQ * EMB;
  f32_to_bf16_kernel<<<(nx + 255) / 256, blk, 0, stream>>>(x, xb, nx);
  int nq = EMB * NH * HD;
  f32_to_bf16_kernel<<<(nq + 255) / 256, blk, 0, stream>>>(Wq, wqb, nq);
  int nk = EMB * NKV * HD;
  f32_to_bf16_kernel<<<(nk + 255) / 256, blk, 0, stream>>>(Wk, wkb, nk);
  f32_to_bf16_kernel<<<(nk + 255) / 256, blk, 0, stream>>>(Wv, wvb, nk);
  int no = (NH * HD) * (NH * HD);
  f32_to_bf16_kernel<<<(no + 255) / 256, blk, 0, stream>>>(Wo, wob, no);

  // projections
  gemm_bf16_wmma<true><<<dim3((NH * HD) / 128, M / 128), blk, 0, stream>>>(
      xb, wqb, nullptr, qb, M, NH * HD, EMB);
  gemm_bf16_wmma<true><<<dim3((NKV * HD) / 128, M / 128), blk, 0, stream>>>(
      xb, wkb, nullptr, kb, M, NKV * HD, EMB);
  gemm_bf16_wmma<true><<<dim3((NKV * HD) / 128, M / 128), blk, 0, stream>>>(
      xb, wvb, nullptr, vb, M, NKV * HD, EMB);

  // RoPE
  int tq = BATCH * SEQ * NH * (HD / 2);
  int tk = BATCH * SEQ * NKV * (HD / 2);
  rope_kernel<<<(tq + 255) / 256, blk, 0, stream>>>(qb, NH, tq);
  rope_kernel<<<(tk + 255) / 256, blk, 0, stream>>>(kb, NKV, tk);

  // attention
  attn_kernel<<<dim3(SEQ / 128, NH, BATCH), blk, 0, stream>>>(qb, kb, vb, yb);

  // output projection (fp32 out + bias)
  gemm_bf16_wmma<false><<<dim3((NH * HD) / 128, M / 128), blk, 0, stream>>>(
      yb, wob, bo, out, M, NH * HD, EMB);
}
